// PatchedMixtralSparseMoeBlock_43533788512804
// MI455X (gfx1250) — compile-verified, matrix operand order checked by reading, not run on hardware
//
#include <hip/hip_runtime.h>

// ---------------------------------------------------------------------------
// Mixtral sparse MoE block for gfx1250 (MI455X), wave32 + WMMA bf16 + TDM.
// T=4096 tokens, D=2048 hidden, F=4096 ffn, E=8 experts, top-2 routing.
// ---------------------------------------------------------------------------

#define NUM_T 4096
#define DIM_D 2048
#define DIM_F 4096
#define NUM_E 8
#define TOPK  2
#define TK    (NUM_T * TOPK)          // 8192 assignments
#define CAP   (TK + NUM_E * 16)       // padded capacity (16-aligned segments)
#define CAPB  (CAP / 16)              // 520 blocks for the MoE kernel

typedef __attribute__((ext_vector_type(16))) __bf16 v16bf;
typedef __attribute__((ext_vector_type(8)))  __bf16 v8bf;
typedef __attribute__((ext_vector_type(8)))  float  v8f;
typedef __attribute__((ext_vector_type(4)))  float  f32x4;
typedef __attribute__((ext_vector_type(4)))  unsigned int u32x4;
typedef __attribute__((ext_vector_type(8)))  int          i32x8;
typedef __attribute__((ext_vector_type(4)))  int          i32x4;

// fp32 -> bf16: plain cast lowers to native v_cvt_pk_bf16_f32.
__device__ __forceinline__ __bf16 f2bf(float f) { return (__bf16)f; }

// A/B fragment from bf16 LDS tile (row-major, `stride` elems per row).
// 16-bit 16x32 layout: lanes<16 hold K {k..k+7, k+16..k+23}; lanes>=16 +8.
__device__ __forceinline__ v16bf frag_lds(const __bf16* base, int row,
                                          int stride, int k, int hi) {
  const __bf16* p = base + row * stride + k + hi;
  v8bf c0 = *(const v8bf*)(p);
  v8bf c1 = *(const v8bf*)(p + 16);
  v16bf r;
#pragma unroll
  for (int i = 0; i < 8; ++i) { r[i] = c0[i]; r[i + 8] = c1[i]; }
  return r;
}

// B fragment from a global fp32 row (one matrix row per lane), converted bf16.
__device__ __forceinline__ v16bf frag_glb(const float* row, int k, int hi) {
  const float* p = row + k + hi;
  f32x4 a0 = *(const f32x4*)(p);
  f32x4 a1 = *(const f32x4*)(p + 4);
  f32x4 b0 = *(const f32x4*)(p + 16);
  f32x4 b1 = *(const f32x4*)(p + 20);
  v16bf r;
#pragma unroll
  for (int i = 0; i < 4; ++i) {
    r[i]      = f2bf(a0[i]);
    r[i + 4]  = f2bf(a1[i]);
    r[i + 8]  = f2bf(b0[i]);
    r[i + 12] = f2bf(b1[i]);
  }
  return r;
}

// silu via fast hardware reciprocal (v_rcp_f32), avoids IEEE div expansion.
__device__ __forceinline__ float silu_mul(float x, float y) {
  return x * __builtin_amdgcn_rcpf(1.f + __expf(-x)) * y;
}

// ---------------------------------------------------------------------------
// Kernel 1: router. One wave32 per token: logits = h . gate_w^T, softmax,
// top-2, renormalize. Writes raw logits to d_out tail, routing info to ws.
// ---------------------------------------------------------------------------
__global__ __launch_bounds__(256) void moe_router_kernel(
    const float* __restrict__ h, const float* __restrict__ gw,
    float* __restrict__ out_logits, int* __restrict__ sel,
    float* __restrict__ selw) {
  const int lane = threadIdx.x & 31;
  const int wv   = threadIdx.x >> 5;
  const int t    = blockIdx.x * 8 + wv;
  if (t >= NUM_T) return;

  float acc[NUM_E];
#pragma unroll
  for (int e = 0; e < NUM_E; ++e) acc[e] = 0.f;

  const float* hrow = h + (size_t)t * DIM_D;
  for (int d = lane; d < DIM_D; d += 32) {
    float x = hrow[d];
#pragma unroll
    for (int e = 0; e < NUM_E; ++e) acc[e] += x * gw[e * DIM_D + d];
  }
#pragma unroll
  for (int e = 0; e < NUM_E; ++e) {
#pragma unroll
    for (int off = 16; off > 0; off >>= 1)
      acc[e] += __shfl_xor(acc[e], off, 32);
  }

  if (lane == 0) {
    float mx = acc[0];
#pragma unroll
    for (int e = 1; e < NUM_E; ++e) mx = fmaxf(mx, acc[e]);
    float p[NUM_E];
#pragma unroll
    for (int e = 0; e < NUM_E; ++e) p[e] = __expf(acc[e] - mx);
    int e0 = 0;
#pragma unroll
    for (int e = 1; e < NUM_E; ++e) if (p[e] > p[e0]) e0 = e;
    int e1 = (e0 == 0) ? 1 : 0;
#pragma unroll
    for (int e = 0; e < NUM_E; ++e)
      if (e != e0 && p[e] > p[e1]) e1 = e;
    float den = p[e0] + p[e1];
    sel[2 * t + 0]  = e0;
    sel[2 * t + 1]  = e1;
    selw[2 * t + 0] = p[e0] / den;
    selw[2 * t + 1] = p[e1] / den;
#pragma unroll
    for (int e = 0; e < NUM_E; ++e) out_logits[t * NUM_E + e] = acc[e];
  }
}

// ---------------------------------------------------------------------------
// Kernel 2: deterministic order-preserving grouping into 16-aligned per-expert
// segments. Single thread (8192 iterations — trivial), no atomics.
// ---------------------------------------------------------------------------
__global__ void moe_group_kernel(const int* __restrict__ sel,
                                 const float* __restrict__ selw,
                                 int* __restrict__ g_tok,
                                 float* __restrict__ g_wt,
                                 int* __restrict__ g_exp,
                                 int* __restrict__ total) {
  if (threadIdx.x != 0 || blockIdx.x != 0) return;
  int cnt[NUM_E];
  for (int e = 0; e < NUM_E; ++e) cnt[e] = 0;
  for (int i = 0; i < TK; ++i) cnt[sel[i]]++;
  int start[NUM_E], off = 0;
  for (int e = 0; e < NUM_E; ++e) {
    start[e] = off;
    off += (cnt[e] + 15) & ~15;
  }
  *total = off;
  int pos[NUM_E];
  for (int e = 0; e < NUM_E; ++e) pos[e] = start[e];
  for (int t = 0; t < NUM_T; ++t) {
    for (int k = 0; k < TOPK; ++k) {
      int e = sel[2 * t + k];
      int i = pos[e]++;
      g_tok[i] = t;
      g_wt[i]  = selw[2 * t + k];
      g_exp[i] = e;
    }
  }
  for (int e = 0; e < NUM_E; ++e) {
    int end = start[e] + ((cnt[e] + 15) & ~15);
    for (int i = start[e] + cnt[e]; i < end; ++i) {
      g_tok[i] = 0; g_wt[i] = 0.f; g_exp[i] = e;
    }
  }
}

// ---------------------------------------------------------------------------
// Kernel 3: fused expert FFN. One block = 16 token-assignments of one expert.
// h rows staged via TDM gather-mode tensor_load_to_lds (16 row indices).
// Phase 1: gated[16,F] = silu(h.w1^T) * (h.w3^T)  (bf16 in 128KB LDS)
// Phase 2: out += (gated . w2^T) * gate_weight    (f32 atomics)
// Per wave iteration: 4 N-tiles share one A fragment (8 WMMA / A-load).
// ---------------------------------------------------------------------------
__global__ __launch_bounds__(256, 1) void moe_ffn_kernel(
    const float* __restrict__ h, const float* __restrict__ w1,
    const float* __restrict__ w2, const float* __restrict__ w3,
    const int* __restrict__ g_tok, const float* __restrict__ g_wt,
    const int* __restrict__ g_exp, const int* __restrict__ total,
    float* __restrict__ out) {
  const int base = blockIdx.x * 16;
  if (base >= *total) return;

  extern __shared__ char smem[];
  __bf16* hA = (__bf16*)smem;                          // [16][2048]  64 KB
  // gA doubles as the 128KB f32 TDM landing zone before phase 1 overwrites it.
  __bf16* gA = (__bf16*)(smem + 16 * DIM_D * 2);       // [16][4096] 128 KB
  int*   tokS = (int*)(smem + 16 * DIM_D * 2 + 16 * DIM_F * 2);
  float* wtS  = (float*)(tokS + 16);

  const int tid  = threadIdx.x;
  const int lane = tid & 31;
  const int wv   = tid >> 5;
  const int m    = lane & 15;              // row / col index inside 16
  const int hi   = (lane & 16) ? 8 : 0;    // K-chunk / M-half selector

  if (tid < 16) {
    tokS[tid] = g_tok[base + tid];
    wtS[tid]  = g_wt[base + tid];
  }
  __syncthreads();
  const int e = g_exp[base];

#if __has_builtin(__builtin_amdgcn_tensor_load_to_lds) && \
    __has_builtin(__builtin_amdgcn_s_wait_tensorcnt)
  // --- TDM gather: DMA the 16 gathered h rows (f32) into LDS -------------
  if (wv == 0) {
    const unsigned lds_off = (unsigned)(size_t)(void*)gA;
    const unsigned long long ga = (unsigned long long)(size_t)h;
    u32x4 g0;
    g0[0] = 0x80000001u;                       // count=1, gather_mode, 16b idx
    g0[1] = lds_off;                           // lds_addr
    g0[2] = (unsigned)(ga & 0xFFFFFFFFu);      // global_addr[31:0]
    g0[3] = (unsigned)((ga >> 32) & 0x01FFFFFFu) | (2u << 30); // addr[56:32], type=2
    i32x8 g1;
    g1[0] = 0x00020000;                        // data_size = 4B
    g1[1] = (int)((unsigned)DIM_D << 16);      // tensor_dim0 = 2048 (lo16)
    g1[2] = (int)((unsigned)NUM_T << 16);      // dim0 hi=0 | tensor_dim1 lo16
    g1[3] = (int)((unsigned)DIM_D << 16);      // dim1 hi=0 | tile_dim0 = 2048
    g1[4] = 16;                                // tile_dim1 = #row indices
    g1[5] = DIM_D;                             // tensor_dim0_stride = 2048
    g1[6] = 0;
    g1[7] = 0;
    i32x4 g2, g3;
#pragma unroll
    for (int j = 0; j < 4; ++j) {
      int lo = __builtin_amdgcn_readfirstlane(tokS[2 * j + 0]);
      int hi16 = __builtin_amdgcn_readfirstlane(tokS[2 * j + 1]);
      g2[j] = (lo & 0xFFFF) | (hi16 << 16);
      int lo2 = __builtin_amdgcn_readfirstlane(tokS[8 + 2 * j + 0]);
      int hi2 = __builtin_amdgcn_readfirstlane(tokS[8 + 2 * j + 1]);
      g3[j] = (lo2 & 0xFFFF) | (hi2 << 16);
    }
    const i32x8 g4 = {0, 0, 0, 0, 0, 0, 0, 0};  // unused trailing group
    __builtin_amdgcn_tensor_load_to_lds(g0, g1, g2, g3, g4, 0);
    __builtin_amdgcn_s_wait_tensorcnt(0);
  }
  __syncthreads();
  // Convert staged f32 rows (in gA region) to bf16 in hA.
  {
    const float* hf = (const float*)gA;
    for (int i = tid; i < 16 * DIM_D; i += 256) hA[i] = f2bf(hf[i]);
  }
  __syncthreads();
#else
  // Fallback: gather+convert straight from global memory.
  for (int i = tid; i < 16 * DIM_D; i += 256) {
    int r = i >> 11;           // /2048
    int d = i & (DIM_D - 1);
    hA[i] = f2bf(h[(size_t)tokS[r] * DIM_D + d]);
  }
  __syncthreads();
#endif

  // ---- Phase 1: gated = silu(h.w1^T) * (h.w3^T) --------------------------
  // 64 columns (4 N-tiles) per wave iteration; A fragment reused 8x per k.
  for (int g = wv; g < DIM_F / 64; g += 8) {
    const int fb = g * 64;
    const float* b1row[4];
    const float* b3row[4];
#pragma unroll
    for (int j = 0; j < 4; ++j) {
      b1row[j] = w1 + ((size_t)e * DIM_F + fb + j * 16 + m) * (size_t)DIM_D;
      b3row[j] = w3 + ((size_t)e * DIM_F + fb + j * 16 + m) * (size_t)DIM_D;
    }
    v8f c1[4] = {}, c3[4] = {};
    for (int kc = 0; kc < DIM_D; kc += 32) {
      v16bf a = frag_lds(hA, m, DIM_D, kc, hi);
#pragma unroll
      for (int j = 0; j < 4; ++j) {
        __builtin_prefetch(b1row[j] + kc + 128, 0, 1);
        v16bf b1 = frag_glb(b1row[j], kc, hi);
        c1[j] = __builtin_amdgcn_wmma_f32_16x16x32_bf16(
            false, a, false, b1, (short)0, c1[j], false, false);
        __builtin_prefetch(b3row[j] + kc + 128, 0, 1);
        v16bf b3 = frag_glb(b3row[j], kc, hi);
        c3[j] = __builtin_amdgcn_wmma_f32_16x16x32_bf16(
            false, a, false, b3, (short)0, c3[j], false, false);
      }
    }
#pragma unroll
    for (int j = 0; j < 4; ++j) {
#pragma unroll
      for (int i = 0; i < 8; ++i) {
        float gv = silu_mul(c1[j][i], c3[j][i]);
        gA[(i + hi) * DIM_F + fb + j * 16 + m] = f2bf(gv);
      }
    }
  }
  __syncthreads();

  // ---- Phase 2: out += (gated . w2^T) * weight ---------------------------
  for (int g = wv; g < DIM_D / 64; g += 8) {
    const int db = g * 64;
    const float* b2row[4];
#pragma unroll
    for (int j = 0; j < 4; ++j)
      b2row[j] = w2 + ((size_t)e * DIM_D + db + j * 16 + m) * (size_t)DIM_F;
    v8f acc[4] = {};
    for (int kc = 0; kc < DIM_F; kc += 32) {
      v16bf a = frag_lds(gA, m, DIM_F, kc, hi);
#pragma unroll
      for (int j = 0; j < 4; ++j) {
        __builtin_prefetch(b2row[j] + kc + 128, 0, 1);
        v16bf b = frag_glb(b2row[j], kc, hi);
        acc[j] = __builtin_amdgcn_wmma_f32_16x16x32_bf16(
            false, a, false, b, (short)0, acc[j], false, false);
      }
    }
#pragma unroll
    for (int j = 0; j < 4; ++j) {
#pragma unroll
      for (int i = 0; i < 8; ++i) {
        int row = i + hi;
        float wgt = wtS[row];
        if (wgt != 0.f)
          atomicAdd(&out[(size_t)tokS[row] * DIM_D + db + j * 16 + m],
                    acc[j][i] * wgt);
      }
    }
  }
}

// ---------------------------------------------------------------------------
// Host-side launch. Inputs: hidden_states, gate_w, w1, w2, w3 (all fp32).
// d_out: final [T*D] floats then router_logits [T*8] floats.
// ---------------------------------------------------------------------------
extern "C" void kernel_launch(void* const* d_in, const int* in_sizes, int n_in,
                              void* d_out, int out_size, void* d_ws,
                              size_t ws_size, hipStream_t stream) {
  const float* h  = (const float*)d_in[0];
  const float* gw = (const float*)d_in[1];
  const float* w1 = (const float*)d_in[2];
  const float* w2 = (const float*)d_in[3];
  const float* w3 = (const float*)d_in[4];

  float* out        = (float*)d_out;
  float* out_logits = out + (size_t)NUM_T * DIM_D;

  // Workspace carve-up (~165 KB total).
  char* ws   = (char*)d_ws;
  int*   sel   = (int*)ws;                    // [T*2]
  float* selw  = (float*)(sel + TK);          // [T*2]
  int*   g_tok = (int*)(selw + TK);           // [CAP]
  float* g_wt  = (float*)(g_tok + CAP);       // [CAP]
  int*   g_exp = (int*)(g_wt + CAP);          // [CAP]
  int*   total = (int*)(g_exp + CAP);         // [1]

  // Zero the accumulated output region (logits region fully overwritten).
  (void)hipMemsetAsync(out, 0, sizeof(float) * (size_t)NUM_T * DIM_D, stream);

  moe_router_kernel<<<NUM_T / 8, 256, 0, stream>>>(h, gw, out_logits, sel,
                                                   selw);
  moe_group_kernel<<<1, 1, 0, stream>>>(sel, selw, g_tok, g_wt, g_exp, total);

  const size_t smem = (size_t)16 * DIM_D * 2   // hA bf16
                    + (size_t)16 * DIM_F * 2   // gated bf16 / f32 TDM stage
                    + 16 * sizeof(int) + 16 * sizeof(float);
  moe_ffn_kernel<<<CAPB, 256, smem, stream>>>(h, w1, w2, w3, g_tok, g_wt,
                                              g_exp, total, out);
}
